// VQVAE_52690658787630
// MI455X (gfx1250) — matrix operand order repaired and checked by reading, not compiled, not run on hardware
//
#include <hip/hip_runtime.h>
#include <hip/hip_bf16.h>

typedef __attribute__((ext_vector_type(16))) __bf16 v16bf;
typedef __attribute__((ext_vector_type(8)))  float  v8f;

// Problem geometry (fixed by the reference)
#define ROWS    65536   // 16*4096
#define DIN     1024    // n_in
#define NOUT    512     // n_out
#define NCODE   2048    // K codes
#define MTILES  4       // 4 x 16 = 64 rows per block
#define MROWS   64

__device__ __forceinline__ unsigned short f32_to_bf16(float f) {
    union { float f; unsigned u; } v; v.f = f;
    unsigned r = (v.u + 0x7FFFu + ((v.u >> 16) & 1u)) >> 16;   // RNE
    return (unsigned short)r;
}

// ---------------------------------------------------------------------------
// Pack W [512,1024] f32 -> bf16 B-fragments: frag(nt,kt) lane l holds
// N = nt*16 + (l&15), K = kt*32 + 8*(l>>4) + j (j<8), kt*32 + 16 + 8*(l>>4) + (j-8) (j>=8)
// ---------------------------------------------------------------------------
__global__ __launch_bounds__(256) void vq_pack_w(const float* __restrict__ W,
                                                 unsigned short* __restrict__ Wp) {
    int t    = blockIdx.x * 256 + threadIdx.x;
    int lane = t & 31;
    int frag = t >> 5;               // nt*32 + kt   (nt<32, kt<32)
    if (frag >= 32 * 32) return;
    int n     = (frag >> 5) * 16 + (lane & 15);
    int kbase = (frag & 31) * 32 + 8 * (lane >> 4);
    const float* src = W + (size_t)n * DIN + kbase;
    unsigned short* dst = Wp + (size_t)frag * 512 + lane * 16;
#pragma unroll
    for (int j = 0; j < 16; ++j) dst[j] = f32_to_bf16(src[j < 8 ? j : j + 8]);
}

// ---------------------------------------------------------------------------
// Pack codebook [2048,512] f32 -> bf16 B-fragments (128 n-tiles x 16 k-tiles)
// ---------------------------------------------------------------------------
__global__ __launch_bounds__(256) void vq_pack_c(const float* __restrict__ C,
                                                 unsigned short* __restrict__ Cp) {
    int t    = blockIdx.x * 256 + threadIdx.x;
    int lane = t & 31;
    int frag = t >> 5;               // nt2*16 + kt  (nt2<128, kt<16)
    if (frag >= 128 * 16) return;
    int n2    = (frag >> 4) * 16 + (lane & 15);
    int kbase = (frag & 15) * 32 + 8 * (lane >> 4);
    const float* src = C + (size_t)n2 * NOUT + kbase;
    unsigned short* dst = Cp + (size_t)frag * 512 + lane * 16;
#pragma unroll
    for (int j = 0; j < 16; ++j) dst[j] = f32_to_bf16(src[j < 8 ? j : j + 8]);
}

// ---------------------------------------------------------------------------
// c_sq[k] = sum_d codebook[k][d]^2
// ---------------------------------------------------------------------------
__global__ __launch_bounds__(256) void vq_csq(const float* __restrict__ C,
                                              float* __restrict__ csq) {
    int n = blockIdx.x * 256 + threadIdx.x;
    if (n >= NCODE) return;
    const float4* p = (const float4*)(C + (size_t)n * NOUT);
    float s = 0.f;
#pragma unroll 8
    for (int i = 0; i < NOUT / 4; ++i) {
        float4 v = p[i];
        s += v.x * v.x + v.y * v.y + v.z * v.z + v.w * v.w;
    }
    csq[n] = s;
}

// ---------------------------------------------------------------------------
// Fused: z = x*W^T + b ; score = |c|^2 - 2 z.c ; idx = argmin ; out = C[idx]
// One block per 64 rows (4 M-tiles). 8 waves (wave32).
// Each wave: GEMM1 owns 4 M x 4 N tiles; GEMM2 owns 4 M x 16 N tiles.
// ---------------------------------------------------------------------------
__global__ __launch_bounds__(256)
__attribute__((amdgpu_waves_per_eu(4)))
void vq_main(const float* __restrict__ x,
             const float* __restrict__ bias,
             const float* __restrict__ Cf32,
             const unsigned short* __restrict__ Wp,
             const unsigned short* __restrict__ Cp,
             const float* __restrict__ csq,
             float* __restrict__ out) {
    // aFrag: one K-chunk (256 of 1024) of the 64-row x tile, A-frag layout:
    //        [mt][kt 0..7][lane][16]  -> 4*8*32*16*2B = 32 KB
    __shared__ unsigned short aFrag[MTILES * 8 * 32 * 16];
    // zFrag: full z tile (64 x 512) in A-frag layout:
    //        [mt][kt 0..15][lane][16] -> 4*16*32*16*2B = 64 KB
    __shared__ unsigned short zFrag[MTILES * 16 * 32 * 16];
    __shared__ float sVal[8][MROWS];
    __shared__ int   sIdx[8][MROWS];
    __shared__ int   bestIdx[MROWS];

    const int tid  = threadIdx.x;
    const int lane = tid & 31;
    const int wv   = tid >> 5;
    const int row0 = blockIdx.x * MROWS;

    // ---- GEMM1: z(64x512) = A(64x1024) x B(1024x512), bf16 WMMA ----------
    v8f acc[MTILES][4];
#pragma unroll
    for (int mt = 0; mt < MTILES; ++mt)
#pragma unroll
        for (int t = 0; t < 4; ++t)
            acc[mt][t] = (v8f){0.f,0.f,0.f,0.f,0.f,0.f,0.f,0.f};

#pragma unroll 1
    for (int kc = 0; kc < 4; ++kc) {          // K chunks of 256
        __syncthreads();                      // aFrag free for restaging
        // stage chunk: 1024 fragment-lanes, 4 per thread
#pragma unroll
        for (int i = 0; i < 4; ++i) {
            int f   = tid * 4 + i;            // 0..1023 = ((mt*8)+kt)*32 + l
            int l   = f & 31;
            int kt  = (f >> 5) & 7;
            int mt  = f >> 8;
            int m   = mt * 16 + (l & 15);
            int h   = l >> 4;
            const float* xr = x + (size_t)(row0 + m) * DIN;
            int k0 = (kc * 8 + kt) * 32 + 8 * h;   // j=0..7  -> K=k0+j
            int k1 = k0 + 16;                      // j=8..15 -> K=k1+(j-8)
            unsigned short tmp[16];
            const float4* p0 = (const float4*)(xr + k0);
            const float4* p1 = (const float4*)(xr + k1);
            float4 a0 = p0[0], a1 = p0[1], b0 = p1[0], b1 = p1[1];
            tmp[0]  = f32_to_bf16(a0.x);  tmp[1]  = f32_to_bf16(a0.y);
            tmp[2]  = f32_to_bf16(a0.z);  tmp[3]  = f32_to_bf16(a0.w);
            tmp[4]  = f32_to_bf16(a1.x);  tmp[5]  = f32_to_bf16(a1.y);
            tmp[6]  = f32_to_bf16(a1.z);  tmp[7]  = f32_to_bf16(a1.w);
            tmp[8]  = f32_to_bf16(b0.x);  tmp[9]  = f32_to_bf16(b0.y);
            tmp[10] = f32_to_bf16(b0.z);  tmp[11] = f32_to_bf16(b0.w);
            tmp[12] = f32_to_bf16(b1.x);  tmp[13] = f32_to_bf16(b1.y);
            tmp[14] = f32_to_bf16(b1.z);  tmp[15] = f32_to_bf16(b1.w);
            unsigned* dst = (unsigned*)(aFrag + (size_t)f * 16);
            const unsigned* srcw = (const unsigned*)tmp;
#pragma unroll
            for (int w = 0; w < 8; ++w) dst[w] = srcw[w];
        }
        __syncthreads();

#pragma unroll 1
        for (int kt = 0; kt < 8; ++kt) {
            v16bf a[MTILES];
#pragma unroll
            for (int mt = 0; mt < MTILES; ++mt)
                a[mt] = *(const v16bf*)(aFrag + ((size_t)(mt * 8 + kt) * 32 + lane) * 16);
#pragma unroll
            for (int t = 0; t < 4; ++t) {
                int nt = wv * 4 + t;
                v16bf bm = *(const v16bf*)(Wp +
                    ((size_t)(nt * 32 + kc * 8 + kt) * 512) + lane * 16);
#pragma unroll
                for (int mt = 0; mt < MTILES; ++mt)
                    acc[mt][t] = __builtin_amdgcn_wmma_f32_16x16x32_bf16(
                        false, a[mt], false, bm, (short)0, acc[mt][t], false, false);
            }
        }
    }

    // ---- add bias, scatter z into GEMM2 A-fragment layout in LDS ----------
    {
        int h = lane >> 4;
#pragma unroll
        for (int t = 0; t < 4; ++t) {
            int nz = (wv * 4 + t) * 16 + (lane & 15);   // z column == GEMM2 K
            float bv = bias[nz];
            int kt2 = nz >> 5;
            int kp  = nz & 31;
            int hp, jp;
            if (kp < 16) { hp = kp >> 3; jp = kp & 7; }
            else         { hp = (kp - 16) >> 3; jp = 8 + ((kp - 16) & 7); }
#pragma unroll
            for (int mt = 0; mt < MTILES; ++mt) {
#pragma unroll
                for (int r = 0; r < 8; ++r) {
                    int m = r + 8 * h;
                    int lanep = hp * 16 + m;
                    zFrag[((size_t)(mt * 16 + kt2) * 32 + lanep) * 16 + jp] =
                        f32_to_bf16(acc[mt][t][r] + bv);
                }
            }
        }
    }
    __syncthreads();

    // ---- GEMM2 + running argmin: score = |c|^2 - 2*(z . c) ---------------
    float bestV[MTILES][8];
    int   bestI[MTILES][8];
#pragma unroll
    for (int mt = 0; mt < MTILES; ++mt)
#pragma unroll
        for (int r = 0; r < 8; ++r) { bestV[mt][r] = 3.4e38f; bestI[mt][r] = 0; }

#pragma unroll 1
    for (int tt = 0; tt < 16; ++tt) {
        int nt2 = wv * 16 + tt;                 // 0..127
        v8f acc2[MTILES];
#pragma unroll
        for (int mt = 0; mt < MTILES; ++mt)
            acc2[mt] = (v8f){0.f,0.f,0.f,0.f,0.f,0.f,0.f,0.f};
#pragma unroll 4
        for (int kt = 0; kt < 16; ++kt) {
            v16bf bm = *(const v16bf*)(Cp + ((size_t)(nt2 * 16 + kt) * 512) + lane * 16);
#pragma unroll
            for (int mt = 0; mt < MTILES; ++mt) {
                v16bf a = *(const v16bf*)(zFrag +
                    ((size_t)(mt * 16 + kt) * 32 + lane) * 16);
                acc2[mt] = __builtin_amdgcn_wmma_f32_16x16x32_bf16(
                    false, a, false, bm, (short)0, acc2[mt], false, false);
            }
        }
        int n2 = nt2 * 16 + (lane & 15);
        float cs = csq[n2];
#pragma unroll
        for (int mt = 0; mt < MTILES; ++mt) {
#pragma unroll
            for (int r = 0; r < 8; ++r) {
                float s = cs - 2.0f * acc2[mt][r];
                if (s < bestV[mt][r] || (s == bestV[mt][r] && n2 < bestI[mt][r])) {
                    bestV[mt][r] = s; bestI[mt][r] = n2;
                }
            }
        }
    }

    // ---- min-reduce across 16 lanes of each half (same m across N) -------
#pragma unroll
    for (int off = 8; off >= 1; off >>= 1) {
#pragma unroll
        for (int mt = 0; mt < MTILES; ++mt) {
#pragma unroll
            for (int r = 0; r < 8; ++r) {
                float ov = __shfl_xor(bestV[mt][r], off, 32);
                int   oi = __shfl_xor(bestI[mt][r], off, 32);
                if (ov < bestV[mt][r] ||
                    (ov == bestV[mt][r] && oi < bestI[mt][r])) {
                    bestV[mt][r] = ov; bestI[mt][r] = oi;
                }
            }
        }
    }
    if ((lane & 15) == 0) {
        int h = lane >> 4;
#pragma unroll
        for (int mt = 0; mt < MTILES; ++mt)
#pragma unroll
            for (int r = 0; r < 8; ++r) {
                sVal[wv][mt * 16 + r + 8 * h] = bestV[mt][r];
                sIdx[wv][mt * 16 + r + 8 * h] = bestI[mt][r];
            }
    }
    __syncthreads();

    // ---- reduce across 8 waves -------------------------------------------
    if (tid < MROWS) {
        float v = sVal[0][tid];
        int   bi = sIdx[0][tid];
#pragma unroll
        for (int w = 1; w < 8; ++w) {
            float ov = sVal[w][tid];
            int   oi = sIdx[w][tid];
            if (ov < v || (ov == v && oi < bi)) { v = ov; bi = oi; }
        }
        bestIdx[tid] = bi;
    }
    __syncthreads();

    // ---- gather q = codebook[idx] (f32) to output ------------------------
    {
        int m  = tid >> 2;                 // 64 rows, 4 threads per row
        int c0 = (tid & 3) * 128;          // 128 floats per thread
        int code = bestIdx[m];
        const float4* src = (const float4*)(Cf32 + (size_t)code * NOUT + c0);
        float4* dst = (float4*)(out + (size_t)(row0 + m) * NOUT + c0);
#pragma unroll
        for (int i = 0; i < 32; ++i) dst[i] = src[i];
    }
}

extern "C" void kernel_launch(void* const* d_in, const int* in_sizes, int n_in,
                              void* d_out, int out_size, void* d_ws, size_t ws_size,
                              hipStream_t stream) {
    const float* x    = (const float*)d_in[0];   // [16,4096,1024]
    const float* W    = (const float*)d_in[1];   // [512,1024]
    const float* b    = (const float*)d_in[2];   // [512]
    const float* C    = (const float*)d_in[3];   // [2048,512]
    float* out        = (float*)d_out;           // [16,4096,512]

    // Workspace layout (bytes): Wp bf16 1MB | Cp bf16 2MB | csq f32 8KB
    unsigned short* Wp = (unsigned short*)d_ws;
    unsigned short* Cp = (unsigned short*)((char*)d_ws + (size_t)1048576);
    float* csq         = (float*)((char*)d_ws + (size_t)3145728);

    vq_pack_w<<<128, 256, 0, stream>>>(W, Wp);   // 32768 threads
    vq_pack_c<<<256, 256, 0, stream>>>(C, Cp);   // 65536 threads
    vq_csq  <<<8,   256, 0, stream>>>(C, csq);   // 2048 threads
    vq_main <<<ROWS / MROWS, 256, 0, stream>>>(x, b, C, Wp, Cp, csq, out);
}
